// _LoRA_qkv_timm_train_73134703116332
// MI455X (gfx1250) — matrix operand order repaired and checked by reading, not compiled, not run on hardware
//
#include <hip/hip_runtime.h>
#include <hip/hip_bf16.h>

typedef __bf16 bf16_t;
typedef __attribute__((ext_vector_type(16))) __bf16 v16bf;
typedef __attribute__((ext_vector_type(8)))  __bf16 v8bf;
typedef __attribute__((ext_vector_type(8)))  float  v8f;
typedef int v4i __attribute__((vector_size(16)));   // matches builtin's b128 pointee

#define D_    768
#define R_    16
#define T_    8
#define NTOK  8192      // B*S
#define E_    2304      // 3*D
#define ZLD   320       // Z width: q cols [0,160), v cols [160,320) (padded)
#define BC_LD 160       // Bcat width: 128 MoE + 16 cur-LoRA + 16 pad
#define SLD   40        // LDS tile row stride (32 + 8 pad) -> conflict-free b128 reads

#if __has_builtin(__builtin_amdgcn_global_load_async_to_lds_b128) && \
    __has_builtin(__builtin_amdgcn_s_wait_asynccnt)
#define USE_ASYNC_LDS 1
#else
#define USE_ASYNC_LDS 0
#endif

#define AS1P(p) ((__attribute__((address_space(1))) v4i*)(p))
#define AS3P(p) ((__attribute__((address_space(3))) v4i*)(p))

// ---------------------------------------------------------------------------
// K1: gate softmax + Frobenius norms -> scales[17] = {sq[8], sv[8], a_cur}
// ---------------------------------------------------------------------------
__global__ __launch_bounds__(256)
void k_scales(const float* __restrict__ Aq, const float* __restrict__ Bq,
              const float* __restrict__ Av, const float* __restrict__ Bv,
              const float* __restrict__ gate, const float* __restrict__ alpha,
              float* __restrict__ scales) {
  __shared__ float red[256];
  __shared__ float norms[32];               // nAq[8], nBq[8], nAv[8], nBv[8]
  const int tid = threadIdx.x;
  const int RD = R_ * D_;
  for (int slot = 0; slot < 32; ++slot) {
    const int which = slot >> 3, t = slot & 7;
    const float* base = (which == 0) ? (Aq + t * RD)
                      : (which == 1) ? (Bq + t * RD)
                      : (which == 2) ? (Av + t * RD)
                      :                (Bv + t * RD);
    float s = 0.f;
    for (int i = tid; i < RD; i += 256) { float v = base[i]; s += v * v; }
    red[tid] = s;
    __syncthreads();
    for (int off = 128; off > 0; off >>= 1) {
      if (tid < off) red[tid] += red[tid + off];
      __syncthreads();
    }
    if (tid == 0) norms[slot] = sqrtf(red[0]);
    __syncthreads();
  }
  if (tid == 0) {
    float l[T_], mx = -1e30f;
    for (int t = 0; t < T_; ++t) { l[t] = gate[t]; mx = fmaxf(mx, l[t]); }
    float den = 0.f;
    for (int t = 0; t < T_; ++t) { l[t] = __expf(l[t] - mx); den += l[t]; }
    for (int t = 0; t < T_; ++t) {
      float beta = l[t] / den;
      scales[t]     = beta / (norms[t]      * norms[8 + t]);
      scales[8 + t] = beta / (norms[16 + t] * norms[24 + t]);
    }
    scales[16] = alpha[T_];
  }
}

// ---------------------------------------------------------------------------
// fp32 -> bf16 convert
// ---------------------------------------------------------------------------
__global__ void k_f2bf(const float* __restrict__ src, bf16_t* __restrict__ dst, int n) {
  int i = blockIdx.x * blockDim.x + threadIdx.x;
  if (i < n) dst[i] = (bf16_t)src[i];
}

// ---------------------------------------------------------------------------
// Acat [320 x 768] bf16: rows 0..127 A_q(t,r), 128..143 la_q, 144..159 pad,
//                        160..287 A_v, 288..303 la_v, 304..319 pad
// ---------------------------------------------------------------------------
__global__ void k_build_acat(const float* __restrict__ Aq, const float* __restrict__ laq,
                             const float* __restrict__ Av, const float* __restrict__ lav,
                             bf16_t* __restrict__ acat) {
  int i = blockIdx.x * blockDim.x + threadIdx.x;
  if (i >= ZLD * D_) return;
  int row = i / D_, col = i % D_;
  float v = 0.f;
  if (row < 128)      v = Aq[row * D_ + col];
  else if (row < 144) v = laq[(row - 128) * D_ + col];
  else if (row < 160) v = 0.f;
  else if (row < 288) v = Av[(row - 160) * D_ + col];
  else if (row < 304) v = lav[(row - 288) * D_ + col];
  acat[i] = (bf16_t)v;
}

// ---------------------------------------------------------------------------
// Bcat_q/Bcat_v [768 x 160] bf16 with beta/norm scale (and a_cur) folded in
// ---------------------------------------------------------------------------
__global__ void k_build_bcat(const float* __restrict__ Bq, const float* __restrict__ lbq,
                             const float* __restrict__ Bv, const float* __restrict__ lbv,
                             const float* __restrict__ scales,
                             bf16_t* __restrict__ bq, bf16_t* __restrict__ bv) {
  int i = blockIdx.x * blockDim.x + threadIdx.x;
  const int n1 = D_ * BC_LD;
  if (i >= 2 * n1) return;
  const int which = (i >= n1) ? 1 : 0;
  const int j = which ? (i - n1) : i;
  const int d = j / BC_LD, c = j % BC_LD;
  const float* B  = which ? Bv : Bq;
  const float* lb = which ? lbv : lbq;
  const float* sc = scales + (which ? 8 : 0);
  float v = 0.f;
  if (c < 128) { int t = c >> 4, r = c & 15; v = B[(t * D_ + d) * R_ + r] * sc[t]; }
  else if (c < 144) { v = lb[d * R_ + (c - 128)] * scales[16]; }
  bf16_t* out = which ? bv : bq;
  out[j] = (bf16_t)v;
}

// ---------------------------------------------------------------------------
// WMMA helpers
// ---------------------------------------------------------------------------
__device__ inline v8f wmma_bf16(v16bf a, v16bf b, v8f c) {
  return __builtin_amdgcn_wmma_f32_16x16x32_bf16(false, a, false, b, (short)0, c, false, false);
}

// Global-direct fragment loaders (used by the small Z GEMM)
__device__ inline v16bf load_frag_a(const bf16_t* __restrict__ base, int ld, int row0, int k0) {
  const int lane = threadIdx.x & 31;
  const int half = lane >> 4, m = lane & 15;
  const bf16_t* p = base + (size_t)(row0 + m) * ld + k0 + half * 8;
  v8bf lo = *(const v8bf*)(p);
  v8bf hi = *(const v8bf*)(p + 16);
  v16bf f;
#pragma unroll
  for (int i = 0; i < 8; ++i) { f[i] = lo[i]; f[i + 8] = hi[i]; }
  return f;
}

__device__ inline v16bf load_frag_b(const bf16_t* __restrict__ base, int ld, int n0, int k0) {
  const int lane = threadIdx.x & 31;
  const int half = lane >> 4, n = lane & 15;
  const bf16_t* p = base + (size_t)(n0 + n) * ld + k0 + half * 8;
  v8bf lo = *(const v8bf*)(p);
  v8bf hi = *(const v8bf*)(p + 16);
  v16bf f;
#pragma unroll
  for (int i = 0; i < 8; ++i) { f[i] = lo[i]; f[i + 8] = hi[i]; }
  return f;
}

// LDS fragment loaders (tile stored row-major with padded stride SLD=40 bf16)
__device__ inline v16bf lds_frag_a(const bf16_t* s, int r0) {
  const int lane = threadIdx.x & 31;
  const int half = lane >> 4, m = lane & 15;
  const bf16_t* p = s + (r0 + m) * SLD + half * 8;
  v8bf lo = *(const v8bf*)(p);
  v8bf hi = *(const v8bf*)(p + 16);
  v16bf f;
#pragma unroll
  for (int i = 0; i < 8; ++i) { f[i] = lo[i]; f[i + 8] = hi[i]; }
  return f;
}

__device__ inline v16bf lds_frag_b(const bf16_t* s, int n0) {
  const int lane = threadIdx.x & 31;
  const int half = lane >> 4, n = lane & 15;
  const bf16_t* p = s + (n0 + n) * SLD + half * 8;
  v8bf lo = *(const v8bf*)(p);
  v8bf hi = *(const v8bf*)(p + 16);
  v16bf f;
#pragma unroll
  for (int i = 0; i < 8; ++i) { f[i] = lo[i]; f[i + 8] = hi[i]; }
  return f;
}

// Cooperative 128x32 bf16 tile copy global -> LDS (per-thread: 32 contiguous B)
__device__ inline void stage_tiles(const bf16_t* __restrict__ gx, int ldx, int xr0, int kx,
                                   const bf16_t* __restrict__ gw, int ldw, int wr0, int kw,
                                   bf16_t* sx, bf16_t* sw) {
  const int tid = threadIdx.x;
  const int row = tid >> 1;
  const int c16 = (tid & 1) << 4;                       // 0 or 16 elements
  const bf16_t* gpx = gx + (size_t)(xr0 + row) * ldx + kx + c16;
  const bf16_t* gpw = gw + (size_t)(wr0 + row) * ldw + kw + c16;
  bf16_t* lpx = sx + row * SLD + c16;
  bf16_t* lpw = sw + row * SLD + c16;
#if USE_ASYNC_LDS
  __builtin_amdgcn_global_load_async_to_lds_b128(AS1P(gpx),     AS3P(lpx),     0, 0);
  __builtin_amdgcn_global_load_async_to_lds_b128(AS1P(gpx + 8), AS3P(lpx + 8), 0, 0);
  __builtin_amdgcn_global_load_async_to_lds_b128(AS1P(gpw),     AS3P(lpw),     0, 0);
  __builtin_amdgcn_global_load_async_to_lds_b128(AS1P(gpw + 8), AS3P(lpw + 8), 0, 0);
#else
  *(v8bf*)(lpx)     = *(const v8bf*)(gpx);
  *(v8bf*)(lpx + 8) = *(const v8bf*)(gpx + 8);
  *(v8bf*)(lpw)     = *(const v8bf*)(gpw);
  *(v8bf*)(lpw + 8) = *(const v8bf*)(gpw + 8);
#endif
}

// ---------------------------------------------------------------------------
// GEMM1: Z[8192 x 320] = x_bf16 @ Acat^T   (bf16 out, f32 accum)
// 8 waves: 4(M) x 2(N); wave tile 32 x 80; block tile 128 x 160; grid (64, 2)
// Tiny GEMM: direct-from-L2 fragments, no staging needed.
// ---------------------------------------------------------------------------
__global__ __launch_bounds__(256)
void k_zgemm(const bf16_t* __restrict__ xb, const bf16_t* __restrict__ acat,
             bf16_t* __restrict__ z) {
  const int w = threadIdx.x >> 5;
  const int wm = w >> 1, wn = w & 1;
  const int row0 = blockIdx.x * 128 + wm * 32;
  const int col0 = blockIdx.y * 160 + wn * 80;
  v8f acc[2][5];
#pragma unroll
  for (int i = 0; i < 2; ++i)
#pragma unroll
    for (int j = 0; j < 5; ++j)
#pragma unroll
      for (int e = 0; e < 8; ++e) acc[i][j][e] = 0.f;

  for (int k0 = 0; k0 < D_; k0 += 32) {
    v16bf a[2], b[5];
#pragma unroll
    for (int i = 0; i < 2; ++i) a[i] = load_frag_a(xb, D_, row0 + i * 16, k0);
#pragma unroll
    for (int j = 0; j < 5; ++j) b[j] = load_frag_b(acat, D_, col0 + j * 16, k0);
#pragma unroll
    for (int i = 0; i < 2; ++i)
#pragma unroll
      for (int j = 0; j < 5; ++j) acc[i][j] = wmma_bf16(a[i], b[j], acc[i][j]);
  }

  const int lane = threadIdx.x & 31;
  const int half = lane >> 4, nidx = lane & 15;
#pragma unroll
  for (int i = 0; i < 2; ++i)
#pragma unroll
    for (int j = 0; j < 5; ++j) {
      const int c = col0 + j * 16 + nidx;
#pragma unroll
      for (int jv = 0; jv < 8; ++jv) {
        const int r = row0 + i * 16 + half * 8 + jv;
        z[(size_t)r * ZLD + c] = (bf16_t)acc[i][j][jv];
      }
    }
}

// ---------------------------------------------------------------------------
// Main GEMM: out[8192 x 2304] = x@W^T + bias (+ Z@Bcat^T in q/v ranges)
// Double-buffered LDS staging via async global->LDS copies (ASYNCcnt pipeline).
// 8 waves: 2(M) x 4(N); wave tile 64 x 32; block tile 128 x 128; grid (64, 18)
// by in [0,6): q correction (K=160); by in [12,18): v correction; else none.
// ---------------------------------------------------------------------------
__global__ __launch_bounds__(256)
void k_main_gemm(const bf16_t* __restrict__ xb, const bf16_t* __restrict__ wb,
                 const float* __restrict__ bias, const bf16_t* __restrict__ z,
                 const bf16_t* __restrict__ bcq, const bf16_t* __restrict__ bcv,
                 float* __restrict__ out) {
  __shared__ __align__(16) bf16_t sX[2][128 * SLD];
  __shared__ __align__(16) bf16_t sW[2][128 * SLD];

  const int w = threadIdx.x >> 5;
  const int wm = w >> 2, wn = w & 3;
  const int rowblk = blockIdx.x * 128;
  const int colblk = blockIdx.y * 128;
  const int by = blockIdx.y;                       // uniform per block
  const bool corr = (by < 6 || by >= 12);
  const bf16_t* bc = (by < 6) ? bcq : bcv;
  const int zoff   = (by < 6) ? 0 : 160;
  const int ecolblk = (by < 6) ? colblk : (colblk - 1536);
  const int NKM = D_ / 32;                         // 24 main stages
  const int nk = corr ? (NKM + BC_LD / 32) : NKM;  // +5 correction stages

  v8f acc[4][2];
#pragma unroll
  for (int i = 0; i < 4; ++i)
#pragma unroll
    for (int j = 0; j < 2; ++j)
#pragma unroll
      for (int e = 0; e < 8; ++e) acc[i][j][e] = 0.f;

  auto stage = [&](int s, int buf) {
    if (s < NKM)
      stage_tiles(xb, D_, rowblk, s * 32, wb, D_, colblk, s * 32, sX[buf], sW[buf]);
    else
      stage_tiles(z, ZLD, rowblk, zoff + (s - NKM) * 32,
                  bc, BC_LD, ecolblk, (s - NKM) * 32, sX[buf], sW[buf]);
  };

  stage(0, 0);
  int buf = 0;
  for (int s = 0; s < nk; ++s) {
    if (s + 1 < nk) stage(s + 1, buf ^ 1);
#if USE_ASYNC_LDS
    if (s + 1 < nk) __builtin_amdgcn_s_wait_asynccnt(4);  // tile-s copies (oldest 4) done
    else            __builtin_amdgcn_s_wait_asynccnt(0);
#endif
    __syncthreads();                                       // publish tile s to all waves
    v16bf a[4], b[2];
#pragma unroll
    for (int i = 0; i < 4; ++i) a[i] = lds_frag_a(sX[buf], wm * 64 + i * 16);
#pragma unroll
    for (int j = 0; j < 2; ++j) b[j] = lds_frag_b(sW[buf], wn * 32 + j * 16);
#pragma unroll
    for (int i = 0; i < 4; ++i)
#pragma unroll
      for (int j = 0; j < 2; ++j) acc[i][j] = wmma_bf16(a[i], b[j], acc[i][j]);
    __syncthreads();                                       // all reads of buf done
    buf ^= 1;
  }

  const int lane = threadIdx.x & 31;
  const int half = lane >> 4, nidx = lane & 15;
#pragma unroll
  for (int j = 0; j < 2; ++j) {
    const int c = colblk + wn * 32 + j * 16 + nidx;
    const float bsum = bias[c];
#pragma unroll
    for (int i = 0; i < 4; ++i)
#pragma unroll
      for (int jv = 0; jv < 8; ++jv) {
        const int r = rowblk + wm * 64 + i * 16 + half * 8 + jv;
        out[(size_t)r * E_ + c] = acc[i][j][jv] + bsum;
      }
  }
}

// ---------------------------------------------------------------------------
extern "C" void kernel_launch(void* const* d_in, const int* in_sizes, int n_in,
                              void* d_out, int out_size, void* d_ws, size_t ws_size,
                              hipStream_t stream) {
  const float* x     = (const float*)d_in[0];
  const float* qkv_w = (const float*)d_in[1];
  const float* qkv_b = (const float*)d_in[2];
  const float* la_q  = (const float*)d_in[3];
  const float* lb_q  = (const float*)d_in[4];
  const float* la_v  = (const float*)d_in[5];
  const float* lb_v  = (const float*)d_in[6];
  const float* A_q   = (const float*)d_in[7];
  const float* B_q   = (const float*)d_in[8];
  const float* A_v   = (const float*)d_in[9];
  const float* B_v   = (const float*)d_in[10];
  const float* gate  = (const float*)d_in[11];
  const float* alpha = (const float*)d_in[12];
  float* out = (float*)d_out;

  // workspace layout (all chunks 256B-aligned)
  char* ws = (char*)d_ws;
  size_t off = 0;
  float*  scales = (float*)(ws + off);  off += 256;
  bf16_t* xb     = (bf16_t*)(ws + off); off += (size_t)NTOK * D_ * 2;   // 12.6 MB
  bf16_t* wb     = (bf16_t*)(ws + off); off += (size_t)E_ * D_ * 2;     //  3.5 MB
  bf16_t* acat   = (bf16_t*)(ws + off); off += (size_t)ZLD * D_ * 2;    //  0.5 MB
  bf16_t* bcq    = (bf16_t*)(ws + off); off += (size_t)D_ * BC_LD * 2;  //  0.25 MB
  bf16_t* bcv    = (bf16_t*)(ws + off); off += (size_t)D_ * BC_LD * 2;  //  0.25 MB
  bf16_t* zbuf   = (bf16_t*)(ws + off); off += (size_t)NTOK * ZLD * 2;  //  5.2 MB

  k_scales<<<1, 256, 0, stream>>>(A_q, B_q, A_v, B_v, gate, alpha, scales);

  {
    int n = NTOK * D_;
    k_f2bf<<<(n + 255) / 256, 256, 0, stream>>>(x, xb, n);
  }
  {
    int n = E_ * D_;
    k_f2bf<<<(n + 255) / 256, 256, 0, stream>>>(qkv_w, wb, n);
  }
  {
    int n = ZLD * D_;
    k_build_acat<<<(n + 255) / 256, 256, 0, stream>>>(A_q, la_q, A_v, la_v, acat);
  }
  {
    int n = 2 * D_ * BC_LD;
    k_build_bcat<<<(n + 255) / 256, 256, 0, stream>>>(B_q, lb_q, B_v, lb_v, scales, bcq, bcv);
  }

  k_zgemm<<<dim3(NTOK / 128, ZLD / 160), 256, 0, stream>>>(xb, acat, zbuf);
  k_main_gemm<<<dim3(NTOK / 128, E_ / 128), 256, 0, stream>>>(xb, wb, qkv_b, zbuf, bcq, bcv, out);
}
